// SageModel_22462678958404
// MI455X (gfx1250) — compile-verified, matrix operand order checked by reading, not run on hardware
//
#include <hip/hip_runtime.h>

typedef __attribute__((ext_vector_type(16))) _Float16 v16h;
typedef __attribute__((ext_vector_type(8)))  _Float16 v8h;
typedef __attribute__((ext_vector_type(8)))  float    v8f;

#define N_NODES 50000
#define N_EDGES 800000

// ---------------- zero fill ----------------
__global__ void zero_f32_kernel(float* __restrict__ p, int n) {
    int i = blockIdx.x * blockDim.x + threadIdx.x;
    if (i < n) p[i] = 0.0f;
}

// ---------------- degree (once) ----------------
__global__ void degree_kernel(const long long* __restrict__ ei, float* __restrict__ cnt) {
    int e = blockIdx.x * blockDim.x + threadIdx.x;
    if (e < N_EDGES) {
        int d = (int)ei[N_EDGES + e];
        atomicAdd(&cnt[d], 1.0f);
    }
}

// ---------------- scatter-add of neighbor features ----------------
// thread t -> edge e = t>>shift, channel c = t & (Cpad-1)
__global__ void scatter_kernel(const float* __restrict__ h, const long long* __restrict__ ei,
                               float* __restrict__ agg, int Cin, int shift, int hStride) {
    int t = blockIdx.x * blockDim.x + threadIdx.x;
    int e = t >> shift;
    int c = t & ((1 << shift) - 1);
    if (e < N_EDGES && c < Cin) {
        int s = (int)ei[e];
        int d = (int)ei[N_EDGES + e];
        atomicAdd(&agg[d * Cin + c], h[(size_t)s * hStride + c]);
    }
}

// ---------------- build f16 feature buffer F = [agg/cnt | h], row-major [N, Kpad] ----------------
__global__ void build_feat_kernel(const float* __restrict__ agg, const float* __restrict__ cnt,
                                  const float* __restrict__ h, _Float16* __restrict__ F,
                                  int Cin, int kshift, int hStride) {
    int t = blockIdx.x * blockDim.x + threadIdx.x;
    int Kpad = 1 << kshift;
    if (t >= N_NODES * Kpad) return;
    int node = t >> kshift;
    int k = t & (Kpad - 1);
    int halfK = Kpad >> 1;
    float v = 0.0f;
    if (k < halfK) {
        if (k < Cin) v = agg[node * Cin + k] / fmaxf(cnt[node], 1.0f);
    } else {
        int c = k - halfK;
        if (c < Cin) v = h[(size_t)node * hStride + c];
    }
    F[t] = (_Float16)v;
}

// ---------------- pack [Wl; Wr] into exact WMMA B-fragment layout + padded bias ----------------
// fragment tid = (((ntile*ksteps + kstep)*32 + lane)*16 + i)
// lane: N = ntile*16 + (lane&15); half = lane>>4
// i<8 : K = kstep*32 + half*8 + i ; i>=8 : K = kstep*32 + 16 + half*8 + (i-8)
__global__ void prep_w_kernel(const float* __restrict__ Wl, const float* __restrict__ bl,
                              const float* __restrict__ Wr, _Float16* __restrict__ Wfrag,
                              float* __restrict__ biasPad, int Cin, int Cout, int Kpad,
                              int ksteps, int nOutTiles) {
    int t = blockIdx.x * blockDim.x + threadIdx.x;
    if (t < 64) biasPad[t] = (t < Cout) ? bl[t] : 0.0f;
    int total = nOutTiles * ksteps * 32 * 16;
    if (t >= total) return;
    int i = t & 15;
    int lane = (t >> 4) & 31;
    int rest = t >> 9;
    int kstep = rest % ksteps;
    int ntile = rest / ksteps;
    int half = lane >> 4, l15 = lane & 15;
    int n = ntile * 16 + l15;
    int k = kstep * 32 + half * 8 + i + ((i >= 8) ? 8 : 0);
    int halfK = Kpad >> 1;
    float w = 0.0f;
    if (k < halfK) {
        if (k < Cin && n < Cout) w = Wl[k * Cout + n];
    } else {
        int c = k - halfK;
        if (c < Cin && n < Cout) w = Wr[c * Cout + n];
    }
    Wfrag[t] = (_Float16)w;
}

// ---------------- WMMA GEMM: out = act(F @ Wcmb + b) ----------------
// One wave per 16-node M-tile; the wave computes ALL NT output tiles so the
// A fragment is loaded once per K-step and reused across NT back-to-back WMMAs.
template <int KSTEPS, int NT, int LAST>
__global__ void sage_wmma_kernel(const _Float16* __restrict__ F, const _Float16* __restrict__ Wfrag,
                                 const float* __restrict__ biasPad, float* __restrict__ out) {
    const int Kpad = KSTEPS * 32;
    int mtile = blockIdx.x;
    int lane  = threadIdx.x;           // 0..31
    int half  = lane >> 4;
    int l15   = lane & 15;
    int row   = mtile * 16 + l15;      // A-matrix row for this lane (M = l15)

    v8f acc[NT];
#pragma unroll
    for (int nt = 0; nt < NT; ++nt) acc[nt] = (v8f){0.f, 0.f, 0.f, 0.f, 0.f, 0.f, 0.f, 0.f};

    const _Float16* fbase = F + (size_t)row * Kpad + half * 8;
    const _Float16* wbase = Wfrag + (size_t)lane * 16;

#pragma unroll
    for (int ks = 0; ks < KSTEPS; ++ks) {
        // A fragment: shared by all NT output tiles
        v8h alo = *(const v8h*)(fbase + ks * 32);
        v8h ahi = *(const v8h*)(fbase + ks * 32 + 16);
        v16h a;
#pragma unroll
        for (int i = 0; i < 8; ++i) { a[i] = alo[i]; a[i + 8] = ahi[i]; }
#pragma unroll
        for (int nt = 0; nt < NT; ++nt) {
            v16h b = *(const v16h*)(wbase + (size_t)(nt * KSTEPS + ks) * 32 * 16);
            acc[nt] = __builtin_amdgcn_wmma_f32_16x16x32_f16(
                /*neg_a=*/false, a, /*neg_b=*/false, b,
                /*c_mod=*/(short)0, acc[nt], /*reuse_a=*/false, /*reuse_b=*/false);
        }
    }

    int m0 = mtile * 16 + half * 8;      // C/D layout: M = r + 8*half
    if (!LAST) {
#pragma unroll
        for (int nt = 0; nt < NT; ++nt) {
            float bv = biasPad[nt * 16 + l15];
#pragma unroll
            for (int r = 0; r < 8; ++r) {
                float v = acc[nt][r] + bv;
                v = fmaxf(v, 0.0f);
                out[(size_t)(m0 + r) * 64 + nt * 16 + l15] = v;
            }
        }
    } else {
        if (l15 == 0) {                   // only output channel 0 is real
            float bv = biasPad[0];
#pragma unroll
            for (int r = 0; r < 8; ++r) {
                float v = acc[0][r] + bv;
                v = 1.0f / (1.0f + __expf(-v));
                out[m0 + r] = v;
            }
        }
    }
}

extern "C" void kernel_launch(void* const* d_in, const int* in_sizes, int n_in,
                              void* d_out, int out_size, void* d_ws, size_t ws_size,
                              hipStream_t stream) {
    (void)in_sizes; (void)n_in; (void)out_size; (void)ws_size;
    const float*     x  = (const float*)d_in[0];
    const long long* ei = (const long long*)d_in[1];      // int64 [2, E]
    const float *Wl[5], *bl[5], *Wr[5];
    for (int i = 0; i < 5; ++i) {
        Wl[i] = (const float*)d_in[2 + 3 * i];
        bl[i] = (const float*)d_in[3 + 3 * i];
        Wr[i] = (const float*)d_in[4 + 3 * i];
    }
    float* out = (float*)d_out;

    // workspace carve-out (256B aligned)
    char* ws = (char*)d_ws;
    size_t off = 0;
    auto alloc = [&](size_t bytes) -> void* {
        void* p = ws + off;
        off = (off + bytes + 255) & ~(size_t)255;
        return p;
    };
    float*    cnt     = (float*)alloc((size_t)N_NODES * 4);
    float*    agg     = (float*)alloc((size_t)N_NODES * 64 * 4);
    float*    hA      = (float*)alloc((size_t)N_NODES * 64 * 4);
    float*    hB      = (float*)alloc((size_t)N_NODES * 64 * 4);
    _Float16* F       = (_Float16*)alloc((size_t)N_NODES * 128 * 2);
    _Float16* Wfrag   = (_Float16*)alloc((size_t)5 * 8192 * 2);   // per layer <= 4*4*32*16 f16
    float*    biasPad = (float*)alloc((size_t)5 * 64 * 4);

    const dim3 b256(256);

    // degree counts (dst-only, shared by all layers)
    zero_f32_kernel<<<(N_NODES + 255) / 256, b256, 0, stream>>>(cnt, N_NODES);
    degree_kernel<<<(N_EDGES + 255) / 256, b256, 0, stream>>>(ei, cnt);

    const int CinArr[5]  = {13, 64, 64, 64, 64};
    const int CoutArr[5] = {64, 64, 64, 64, 1};
    const int KpadArr[5] = {32, 128, 128, 128, 128};

    const float* hcur = x;
    int hStride = 13;
    for (int L = 0; L < 5; ++L) {
        int Cin = CinArr[L], Cout = CoutArr[L], Kpad = KpadArr[L];
        int ksteps = Kpad / 32;
        int nOutTiles = (Cout == 1) ? 1 : 4;
        _Float16* wf = Wfrag + (size_t)L * 8192;
        float*    bp = biasPad + L * 64;

        int wtot = nOutTiles * ksteps * 32 * 16;
        int prepThreads = wtot > 64 ? wtot : 64;
        prep_w_kernel<<<(prepThreads + 255) / 256, b256, 0, stream>>>(
            Wl[L], bl[L], Wr[L], wf, bp, Cin, Cout, Kpad, ksteps, nOutTiles);

        zero_f32_kernel<<<(N_NODES * Cin + 255) / 256, b256, 0, stream>>>(agg, N_NODES * Cin);

        int shift = (Cin == 13) ? 4 : 6;          // Cpad = 16 or 64
        int sthreads = N_EDGES << shift;
        scatter_kernel<<<(sthreads + 255) / 256, b256, 0, stream>>>(hcur, ei, agg, Cin, shift, hStride);

        int kshift = (Kpad == 32) ? 5 : 7;
        int btotal = N_NODES * Kpad;
        build_feat_kernel<<<(btotal + 255) / 256, b256, 0, stream>>>(agg, cnt, hcur, F, Cin, kshift, hStride);

        float* hn = (L == 4) ? out : ((L & 1) ? hB : hA);
        dim3 grid(N_NODES / 16);                  // 50000 == 16*3125 exactly
        if (L == 4) {
            sage_wmma_kernel<4, 1, 1><<<grid, dim3(32), 0, stream>>>(F, wf, bp, hn);
        } else if (ksteps == 1) {
            sage_wmma_kernel<1, 4, 0><<<grid, dim3(32), 0, stream>>>(F, wf, bp, hn);
        } else {
            sage_wmma_kernel<4, 4, 0><<<grid, dim3(32), 0, stream>>>(F, wf, bp, hn);
        }
        hcur = hn;
        hStride = 64;
    }
}
